// BiAttention_65146063946197
// MI455X (gfx1250) — matrix operand order repaired and checked by reading, not compiled, not run on hardware
//
#include <hip/hip_runtime.h>
#include <hip/hip_bf16.h>

#define D_MODEL 1024
#define NH      16
#define HD      64
#define B_SZ    2
#define T_SEQ   2048
#define BT      (B_SZ * T_SEQ)   // 4096 rows

typedef __bf16 bf16x16 __attribute__((ext_vector_type(16)));
typedef float  f32x8   __attribute__((ext_vector_type(8)));

union ABf {
    bf16x16        v;
    unsigned int   u[8];
    unsigned short s[16];
};

__device__ __forceinline__ unsigned short f2bf(float f) {
    unsigned int u = __float_as_uint(f);
    unsigned int r = (u + 0x7FFFu + ((u >> 16) & 1u)) >> 16;   // RNE
    return (unsigned short)r;
}

// 16-lane max reduction entirely on the VALU via DPP row ops (no LDS/bpermute).
__device__ __forceinline__ float dpp_max16(float x) {
    int xi;
    xi = __builtin_amdgcn_update_dpp(0, __float_as_int(x), 0xB1,  0xF, 0xF, true); // quad_perm [1,0,3,2]
    x  = fmaxf(x, __int_as_float(xi));
    xi = __builtin_amdgcn_update_dpp(0, __float_as_int(x), 0x4E,  0xF, 0xF, true); // quad_perm [2,3,0,1]
    x  = fmaxf(x, __int_as_float(xi));
    xi = __builtin_amdgcn_update_dpp(0, __float_as_int(x), 0x141, 0xF, 0xF, true); // row_half_mirror
    x  = fmaxf(x, __int_as_float(xi));
    xi = __builtin_amdgcn_update_dpp(0, __float_as_int(x), 0x140, 0xF, 0xF, true); // row_mirror
    x  = fmaxf(x, __int_as_float(xi));
    return x;
}

// ---------------------------------------------------------------- RMSNorm -> bf16
__global__ __launch_bounds__(256)
void rmsnorm_kernel(const float* __restrict__ x, const float* __restrict__ w,
                    unsigned short* __restrict__ hout) {
    int row = blockIdx.x;
    int t   = threadIdx.x;
    const float* xr = x + (size_t)row * D_MODEL;
    float vals[4];
    float ss = 0.f;
#pragma unroll
    for (int i = 0; i < 4; ++i) { vals[i] = xr[t + i * 256]; ss += vals[i] * vals[i]; }
    __shared__ float red[256];
    red[t] = ss;
    __syncthreads();
    for (int s = 128; s > 0; s >>= 1) {
        if (t < s) red[t] += red[t + s];
        __syncthreads();
    }
    float inv = rsqrtf(red[0] * (1.0f / D_MODEL) + 1e-5f);
#pragma unroll
    for (int i = 0; i < 4; ++i) {
        int c = t + i * 256;
        hout[(size_t)row * D_MODEL + c] = f2bf(vals[i] * inv * w[c]);
    }
}

// ---------------------------------------------------------------- tiled transpose f32 -> bf16
// in [R][C] f32  ->  outT [C][R] bf16.  grid (C/32, R/32), block 256 (32x8).
__global__ __launch_bounds__(256)
void transpose_tobf16_kernel(const float* __restrict__ in, unsigned short* __restrict__ outT,
                             int R, int C) {
    __shared__ float ls[32][33];
    int tx = threadIdx.x & 31, ty = threadIdx.x >> 5;
    int c0 = blockIdx.x * 32, r0 = blockIdx.y * 32;
#pragma unroll
    for (int i = 0; i < 4; ++i)
        ls[ty + i * 8][tx] = in[(size_t)(r0 + ty + i * 8) * C + c0 + tx];
    __syncthreads();
#pragma unroll
    for (int i = 0; i < 4; ++i)
        outT[(size_t)(c0 + ty + i * 8) * R + r0 + tx] = f2bf(ls[tx][ty + i * 8]);
}

// ---------------------------------------------------------------- bf16 WMMA GEMM (B pre-transposed)
// C[M,N] = A[M,K] @ Bw[N,K]^T (+ Res). Block tile 128x64, 8 waves x (32x32), K-step 32.
template <bool ADDRES>
__global__ __launch_bounds__(256)
void gemm_bf16_nt_kernel(const unsigned short* __restrict__ A,
                         const unsigned short* __restrict__ Bw,
                         const float* __restrict__ Res,
                         float* __restrict__ C,
                         int M, int N, int K) {
    __shared__ unsigned short lsA[128 * 32];
    __shared__ unsigned short lsB[64 * 32];
    int bm   = blockIdx.x * 128;
    int bn   = blockIdx.y * 64;
    int t    = threadIdx.x;
    int lane = t & 31;
    int wave = t >> 5;
    int wm   = (wave >> 1) * 32;
    int wn   = (wave & 1) * 32;
    int hlf  = lane >> 4;
    int lr   = lane & 15;

    int am  = t >> 1, ako = (t & 1) * 16;
    int bnr = t >> 2, bko = (t & 3) * 8;
    const unsigned short* ap0 = A + (size_t)(bm + am) * K + ako;
    const unsigned short* bp0 = Bw + (size_t)(bn + bnr) * K + bko;

    f32x8 acc[2][2];
#pragma unroll
    for (int i = 0; i < 2; ++i)
#pragma unroll
        for (int j = 0; j < 2; ++j) acc[i][j] = 0.f;

    uint4 ra0 = *(const uint4*)(ap0);
    uint4 ra1 = *(const uint4*)(ap0 + 8);
    uint4 rb  = *(const uint4*)(bp0);

    for (int k0 = 0; k0 < K; k0 += 32) {
        *(uint4*)&lsA[am * 32 + ako]     = ra0;
        *(uint4*)&lsA[am * 32 + ako + 8] = ra1;
        *(uint4*)&lsB[bnr * 32 + bko]    = rb;
        __syncthreads();

        if (k0 + 32 < K) {          // pipeline: issue next tile's global loads now
            ra0 = *(const uint4*)(ap0 + k0 + 32);
            ra1 = *(const uint4*)(ap0 + k0 + 40);
            rb  = *(const uint4*)(bp0 + k0 + 32);
            if (k0 + 64 < K) {      // gfx1250 prefetch two tiles ahead
                __builtin_prefetch(ap0 + k0 + 64, 0, 1);
                __builtin_prefetch(bp0 + k0 + 64, 0, 1);
            }
        }

        ABf a[2], b[2];
#pragma unroll
        for (int ti = 0; ti < 2; ++ti) {
            int m = wm + ti * 16 + lr;
#pragma unroll
            for (int v = 0; v < 8; ++v) {
                int kk = ((v >> 2) << 4) + (hlf << 3) + ((v & 3) << 1);
                a[ti].u[v] = *(const unsigned int*)&lsA[m * 32 + kk];
            }
        }
#pragma unroll
        for (int tj = 0; tj < 2; ++tj) {
            int n = wn + tj * 16 + lr;
#pragma unroll
            for (int v = 0; v < 8; ++v) {
                int kk = ((v >> 2) << 4) + (hlf << 3) + ((v & 3) << 1);
                b[tj].u[v] = *(const unsigned int*)&lsB[n * 32 + kk];
            }
        }
#pragma unroll
        for (int ti = 0; ti < 2; ++ti)
#pragma unroll
            for (int tj = 0; tj < 2; ++tj)
                acc[ti][tj] = __builtin_amdgcn_wmma_f32_16x16x32_bf16(
                    false, a[ti].v, false, b[tj].v, (short)0, acc[ti][tj], false, false);
        __syncthreads();
    }

#pragma unroll
    for (int ti = 0; ti < 2; ++ti)
#pragma unroll
        for (int tj = 0; tj < 2; ++tj) {
            int m0 = bm + wm + ti * 16 + hlf * 8;
            int n  = bn + wn + tj * 16 + lr;
            size_t base = (size_t)m0 * N + n;
            float* cp = C + base;
            const float* rp = Res + base;
#pragma unroll
            for (int r = 0; r < 8; ++r) {
                float val = acc[ti][tj][r];
                if (ADDRES) val += rp[r * (size_t)N];
                cp[r * (size_t)N] = val;
            }
        }
}

// ---------------------------------------------------------------- RoPE -> Q/K bf16 [B*H, T, 64]
__global__ __launch_bounds__(256)
void rope_qk_kernel(const float* __restrict__ qkv,
                    unsigned short* __restrict__ qb,
                    unsigned short* __restrict__ kb) {
    int idx  = blockIdx.x * 256 + threadIdx.x;   // over BT * NH * HD
    int i    = idx & 63;
    int h    = (idx >> 6) & 15;
    int bt   = idx >> 10;
    int tpos = bt & (T_SEQ - 1);
    int b    = bt >> 11;
    size_t base = (size_t)bt * (3 * D_MODEL);

    float q = qkv[base + 0 * D_MODEL + h * 64 + i];
    float k = qkv[base + 1 * D_MODEL + h * 64 + i];

    int   j    = i & 31;
    float invf = __powf(10000.0f, -(float)(2 * j) * (1.0f / 64.0f));
    float ang  = (float)tpos * invf;
    float sn, cs;
    sincosf(ang, &sn, &cs);

    float qr, kr;
    if (i < 32) {
        qr = -qkv[base + 0 * D_MODEL + h * 64 + i + 32];
        kr = -qkv[base + 1 * D_MODEL + h * 64 + i + 32];
    } else {
        qr = qkv[base + 0 * D_MODEL + h * 64 + i - 32];
        kr = qkv[base + 1 * D_MODEL + h * 64 + i - 32];
    }
    size_t o = ((size_t)(b * NH + h) * T_SEQ + tpos) * 64 + i;
    qb[o] = f2bf(q * cs + qr * sn);
    kb[o] = f2bf(k * cs + kr * sn);
}

// ---------------------------------------------------------------- V -> bf16, transposed [B*H, 64, T]
__global__ __launch_bounds__(256)
void v_transpose_kernel(const float* __restrict__ qkv, unsigned short* __restrict__ vT) {
    __shared__ unsigned short ls[32][72];   // [tpos][i], padded
    int bh = blockIdx.x;
    int b  = bh >> 4, h = bh & 15;
    int t0 = blockIdx.y * 32;
    int t  = threadIdx.x;

    int tp = t >> 3, i0 = (t & 7) * 8;
    const float* src = qkv + (size_t)(b * T_SEQ + t0 + tp) * (3 * D_MODEL) + 2 * D_MODEL + h * 64 + i0;
#pragma unroll
    for (int j = 0; j < 8; ++j) ls[tp][i0 + j] = f2bf(src[j]);
    __syncthreads();

    int i = t >> 2, tpo = (t & 3) * 8;
    union { unsigned short s[8]; uint4 q; } u;
#pragma unroll
    for (int j = 0; j < 8; ++j) u.s[j] = ls[tpo + j][i];
    *(uint4*)&vT[((size_t)bh * 64 + i) * T_SEQ + t0 + tpo] = u.q;
}

// ---------------------------------------------------------------- Flash attention
// Block: 128 thr = 4 waves, each wave 32 query rows; KV chunks of 32 staged in LDS per block.
// Row-sum tracked by an extra "P @ ones" WMMA accumulator tile; row-max via DPP (pure VALU).
__global__ __launch_bounds__(128)
void flash_attn_kernel(const unsigned short* __restrict__ Qb,
                       const unsigned short* __restrict__ Kb,
                       const unsigned short* __restrict__ VTb,
                       unsigned short* __restrict__ Ob) {
    __shared__ unsigned short lsK[32 * 64];     // [kv_local][hd]
    __shared__ unsigned short lsVT[64 * 32];    // [hd][kv_local]
    __shared__ unsigned short lsP[4][32 * 32];  // per-wave P scratch
    int bh   = blockIdx.x;
    int b    = bh >> 4;
    int h    = bh & 15;
    int t    = threadIdx.x;
    int wave = t >> 5;
    int lane = t & 31;
    int hlf  = lane >> 4;
    int lr   = lane & 15;
    int q0   = blockIdx.y * 128 + wave * 32;

    const unsigned short* Q  = Qb  + (size_t)bh * T_SEQ * 64;
    const unsigned short* K  = Kb  + (size_t)bh * T_SEQ * 64;
    const unsigned short* Vt = VTb + (size_t)bh * 64 * T_SEQ;

    // constant all-ones B fragment (bf16 1.0 pairs) for the row-sum column
    ABf ones;
#pragma unroll
    for (int v = 0; v < 8; ++v) ones.u[v] = 0x3F803F80u;

    // Q fragments: 2 row-tiles x 2 K-halves (head dim 64 = 2 WMMA K-steps)
    ABf aq[2][2];
#pragma unroll
    for (int ti = 0; ti < 2; ++ti)
#pragma unroll
        for (int kh = 0; kh < 2; ++kh) {
            int m = q0 + ti * 16 + lr;
#pragma unroll
            for (int v = 0; v < 8; ++v) {
                int kk = kh * 32 + ((v >> 2) << 4) + (hlf << 3) + ((v & 3) << 1);
                aq[ti][kh].u[v] = *(const unsigned int*)&Q[(size_t)m * 64 + kk];
            }
        }

    // accO[ti][0..3] = output head-dim tiles; accO[ti][4] = running row-sum (P @ ones)
    f32x8 accO[2][5];
#pragma unroll
    for (int ti = 0; ti < 2; ++ti)
#pragma unroll
        for (int j = 0; j < 5; ++j) accO[ti][j] = 0.f;
    float mrun[2][8];
#pragma unroll
    for (int ti = 0; ti < 2; ++ti)
#pragma unroll
        for (int r = 0; r < 8; ++r) mrun[ti][r] = -3.0e38f;

    const float scale = 0.125f;   // 1/sqrt(64)

    for (int kv = 0; kv < T_SEQ; kv += 32) {
        __syncthreads();   // previous chunk's LDS reads complete
        {   // stage K chunk: 32x64
            int r   = t >> 2;
            int off = (t & 3) * 16;
            const unsigned short* src = K + (size_t)(kv + r) * 64 + off;
            *(uint4*)&lsK[r * 64 + off]     = *(const uint4*)src;
            *(uint4*)&lsK[r * 64 + off + 8] = *(const uint4*)(src + 8);
        }
        {   // stage V^T chunk: 64x32
            int n   = t >> 1;
            int off = (t & 1) * 16;
            const unsigned short* src = Vt + (size_t)n * T_SEQ + kv + off;
            *(uint4*)&lsVT[n * 32 + off]     = *(const uint4*)src;
            *(uint4*)&lsVT[n * 32 + off + 8] = *(const uint4*)(src + 8);
        }
        __syncthreads();

        // S = Q @ K^T for a 32x32 chunk
        f32x8 s[2][2];
#pragma unroll
        for (int ti = 0; ti < 2; ++ti)
#pragma unroll
            for (int tj = 0; tj < 2; ++tj) s[ti][tj] = 0.f;

#pragma unroll
        for (int tj = 0; tj < 2; ++tj) {
#pragma unroll
            for (int kh = 0; kh < 2; ++kh) {
                ABf bk;
#pragma unroll
                for (int v = 0; v < 8; ++v) {
                    int kk = kh * 32 + ((v >> 2) << 4) + (hlf << 3) + ((v & 3) << 1);
                    bk.u[v] = *(const unsigned int*)&lsK[(tj * 16 + lr) * 64 + kk];
                }
#pragma unroll
                for (int ti = 0; ti < 2; ++ti)
                    s[ti][tj] = __builtin_amdgcn_wmma_f32_16x16x32_bf16(
                        false, aq[ti][kh].v, false, bk.v, (short)0, s[ti][tj], false, false);
            }
        }

        // online softmax: row-max via DPP; row-sum deferred to the ones-column WMMA
#pragma unroll
        for (int ti = 0; ti < 2; ++ti) {
#pragma unroll
            for (int r = 0; r < 8; ++r) {
                float mx = dpp_max16(fmaxf(s[ti][0][r], s[ti][1][r])) * scale;
                float mold = mrun[ti][r];
                float mnew = fmaxf(mold, mx);
                float corr = __expf(mold - mnew);
                mrun[ti][r] = mnew;
                float p0 = __expf(s[ti][0][r] * scale - mnew);
                float p1 = __expf(s[ti][1][r] * scale - mnew);
#pragma unroll
                for (int j = 0; j < 5; ++j) accO[ti][j][r] *= corr;   // includes row-sum tile
                int row = ti * 16 + hlf * 8 + r;
                lsP[wave][row * 32 + lr]      = f2bf(p0);
                lsP[wave][row * 32 + 16 + lr] = f2bf(p1);
            }
        }
        asm volatile("s_wait_dscnt 0" ::: "memory");   // in-wave P re-layout visible

        // O += P @ V_chunk ; row-sum += P @ ones
        ABf ap[2];
#pragma unroll
        for (int ti = 0; ti < 2; ++ti) {
            int m = ti * 16 + lr;
#pragma unroll
            for (int v = 0; v < 8; ++v) {
                int kk = ((v >> 2) << 4) + (hlf << 3) + ((v & 3) << 1);
                ap[ti].u[v] = *(const unsigned int*)&lsP[wave][m * 32 + kk];
            }
        }
#pragma unroll
        for (int j = 0; j < 4; ++j) {
            ABf bv;
#pragma unroll
            for (int v = 0; v < 8; ++v) {
                int kk = ((v >> 2) << 4) + (hlf << 3) + ((v & 3) << 1);
                bv.u[v] = *(const unsigned int*)&lsVT[(j * 16 + lr) * 32 + kk];
            }
#pragma unroll
            for (int ti = 0; ti < 2; ++ti)
                accO[ti][j] = __builtin_amdgcn_wmma_f32_16x16x32_bf16(
                    false, ap[ti].v, false, bv.v, (short)0, accO[ti][j], false, false);
        }
#pragma unroll
        for (int ti = 0; ti < 2; ++ti)
            accO[ti][4] = __builtin_amdgcn_wmma_f32_16x16x32_bf16(
                false, ap[ti].v, false, ones.v, (short)0, accO[ti][4], false, false);
    }

    // epilogue: O / rowsum -> bf16, head-interleaved [B*T, D_MODEL]
#pragma unroll
    for (int ti = 0; ti < 2; ++ti)
#pragma unroll
        for (int j = 0; j < 4; ++j) {
            size_t obase = (size_t)(b * T_SEQ + q0 + ti * 16 + hlf * 8) * D_MODEL + h * 64 + j * 16 + lr;
#pragma unroll
            for (int r = 0; r < 8; ++r) {
                float o = accO[ti][j][r] / accO[ti][4][r];
                Ob[obase + (size_t)r * D_MODEL] = f2bf(o);
            }
        }
}

// ---------------------------------------------------------------- host launcher
extern "C" void kernel_launch(void* const* d_in, const int* in_sizes, int n_in,
                              void* d_out, int out_size, void* d_ws, size_t ws_size,
                              hipStream_t stream) {
    const float* x      = (const float*)d_in[0];
    const float* norm_w = (const float*)d_in[1];
    const float* w_qkv  = (const float*)d_in[2];
    const float* w_out  = (const float*)d_in[3];
    float* out          = (float*)d_out;

    char* ws = (char*)d_ws;
    size_t off = 0;
    auto alloc = [&](size_t bytes) { char* p = ws + off; off += (bytes + 255) & ~(size_t)255; return p; };

    unsigned short* h_bf   = (unsigned short*)alloc((size_t)BT * D_MODEL * 2);
    unsigned short* wqkvT  = (unsigned short*)alloc((size_t)D_MODEL * 3 * D_MODEL * 2);
    unsigned short* woutT  = (unsigned short*)alloc((size_t)D_MODEL * D_MODEL * 2);
    float*          qkv_f  = (float*)         alloc((size_t)BT * 3 * D_MODEL * 4);
    unsigned short* q_bf   = (unsigned short*)alloc((size_t)BT * D_MODEL * 2);
    unsigned short* k_bf   = (unsigned short*)alloc((size_t)BT * D_MODEL * 2);
    unsigned short* vT_bf  = (unsigned short*)alloc((size_t)BT * D_MODEL * 2);
    unsigned short* ao_bf  = (unsigned short*)alloc((size_t)BT * D_MODEL * 2);

    rmsnorm_kernel<<<BT, 256, 0, stream>>>(x, norm_w, h_bf);

    transpose_tobf16_kernel<<<dim3(3 * D_MODEL / 32, D_MODEL / 32), 256, 0, stream>>>(
        w_qkv, wqkvT, D_MODEL, 3 * D_MODEL);
    transpose_tobf16_kernel<<<dim3(D_MODEL / 32, D_MODEL / 32), 256, 0, stream>>>(
        w_out, woutT, D_MODEL, D_MODEL);

    gemm_bf16_nt_kernel<false><<<dim3(BT / 128, 3 * D_MODEL / 64), 256, 0, stream>>>(
        h_bf, wqkvT, nullptr, qkv_f, BT, 3 * D_MODEL, D_MODEL);

    rope_qk_kernel<<<(BT * D_MODEL) / 256, 256, 0, stream>>>(qkv_f, q_bf, k_bf);
    v_transpose_kernel<<<dim3(B_SZ * NH, T_SEQ / 32), 256, 0, stream>>>(qkv_f, vT_bf);

    flash_attn_kernel<<<dim3(B_SZ * NH, T_SEQ / 128), 128, 0, stream>>>(q_bf, k_bf, vT_bf, ao_bf);

    gemm_bf16_nt_kernel<true><<<dim3(BT / 128, D_MODEL / 64), 256, 0, stream>>>(
        ao_bf, woutT, x, out, BT, D_MODEL, D_MODEL);
}
